// DecoderRNNT_52072183497488
// MI455X (gfx1250) — matrix operand order repaired and checked by reading, not compiled, not run on hardware
//
#include <hip/hip_runtime.h>
#include <hip/hip_bf16.h>
#include <math.h>

// ---------------------------------------------------------------------------
// RNN-T decoder + joint network for MI455X (gfx1250), bf16 WMMA everywhere.
// Dims: B=8, T=150, U=40, EPROJS=512, DUNITS=1024, EMBED=512, JOINT=640, ODIM=1024
// ---------------------------------------------------------------------------

typedef __bf16 bf16_t;
typedef bf16_t v16bf __attribute__((ext_vector_type(16)));
typedef bf16_t v8bf  __attribute__((ext_vector_type(8)));
typedef float  v8f   __attribute__((ext_vector_type(8)));

#define BB      8
#define TT      150
#define UU      40
#define EPROJS  512
#define DUNITS  1024
#define EMBED   512
#define JOINTD  640
#define ODIM    1024
#define JKT     (JOINTD / 32)   // 20 K-tiles in the joint GEMM

__device__ __forceinline__ float fast_tanh(float x) {
#if __has_builtin(__builtin_amdgcn_tanhf)
  return __builtin_amdgcn_tanhf(x);        // V_TANH_F32 (TRANS op on gfx1250)
#else
  return tanhf(x);
#endif
}
__device__ __forceinline__ float fast_sigmoid(float x) {
  return 1.0f / (1.0f + __expf(-x));
}

__device__ __forceinline__ v16bf wmma_bf16(v16bf a, v16bf b, v8f& acc) {
  acc = __builtin_amdgcn_wmma_f32_16x16x32_bf16(false, a, false, b,
                                                (short)0, acc, false, false);
  return a;
}

// ---------------------------------------------------------------------------
// Pack torch-style weight W (N x K, row-major, fp32) into bf16 B-fragments for
// D = A @ W^T.  Fragment (kt,nt): lane L holds 16 contiguous bf16 where
//   n = nt*16 + (L&15),  k = kt*32 + (L<16 ? j : 16+j)      (contiguous k order, same as A)
// dst index: (((kt+ktOff)*NT + nt)*32 + lane)*16 + j  -> one 32B load per lane.
// ---------------------------------------------------------------------------
__global__ void pack_b_kernel(const float* __restrict__ W, bf16_t* __restrict__ dst,
                              int K, int KtLocal, int NT, int ktOff) {
  int e = blockIdx.x * blockDim.x + threadIdx.x;
  int total = KtLocal * NT * 512;
  if (e >= total) return;
  int kt  = e / (NT * 512);
  int rem = e - kt * (NT * 512);
  int nt  = rem >> 9;
  int q   = rem & 511;
  int ln  = q >> 4;
  int j   = q & 15;
  int n   = nt * 16 + (ln & 15);
  int k   = kt * 32 + ((ln < 16) ? j : 16 + j);
  dst[((((size_t)(kt + ktOff)) * NT + nt) * 32 + ln) * 16 + j] =
      (bf16_t)W[(size_t)n * K + k];
}

// ---------------------------------------------------------------------------
// Generic WMMA GEMM:  C[M x (NT*16)] = concatK(A1,A2) @ Bpack + bias
// One wave per 16x16 tile; 4 waves / block (block covers 4 N-tiles).
// A1/A2 are bf16 row-major; K split at k1t*32 (multiple of 32 by construction).
// The K loop is split into two straight-line loops (no per-iteration pointer
// select) so the steady state is 4x global_load_b128 + v_wmma per tile.
// A fragment: lane L -> row (L&15), k = kt*32 + (L<16? {0..15}
// : {16..31})  => two contiguous 16B loads per lane.
// ---------------------------------------------------------------------------
__device__ __forceinline__ v16bf load_a_frag(const bf16_t* __restrict__ arow, int kb) {
  v8bf alo = *(const v8bf*)(arow + kb);
  v8bf ahi = *(const v8bf*)(arow + kb + 8);
  v16bf a;
#pragma unroll
  for (int j = 0; j < 8; ++j) { a[j] = alo[j]; a[j + 8] = ahi[j]; }
  return a;
}

__global__ __launch_bounds__(128) void gemm_wmma_bf16(
    const bf16_t* __restrict__ A1, int lda1, int k1t,
    const bf16_t* __restrict__ A2, int lda2, int k2t,
    const bf16_t* __restrict__ Bp, int NT,
    const float*  __restrict__ bias,
    float* __restrict__ C, int ldc, int M) {
  const int lane  = threadIdx.x & 31;
  const int wid   = threadIdx.x >> 5;
  const int mt    = blockIdx.x;
  const int nt    = blockIdx.y * 4 + wid;
  const int laneM = lane & 15;
  const int hi    = lane >> 4;
  const int koff  = hi ? 16 : 0;

  int rowA = mt * 16 + laneM;
  if (rowA > M - 1) rowA = M - 1;       // clamp pad rows (loads stay in-bounds)

  v8f acc = {0.f, 0.f, 0.f, 0.f, 0.f, 0.f, 0.f, 0.f};

  // B fragment stream for this wave: consecutive kt are NT*512 elements apart.
  const bf16_t* bptr = Bp + ((size_t)nt * 32 + lane) * 16;
  const size_t  bstep = (size_t)NT * 512;

  const bf16_t* arow1 = A1 + (size_t)rowA * lda1;
  for (int kt = 0; kt < k1t; ++kt) {
    v16bf a = load_a_frag(arow1, kt * 32 + koff);
    v16bf b = *(const v16bf*)(bptr);
    bptr += bstep;
    acc = __builtin_amdgcn_wmma_f32_16x16x32_bf16(false, a, false, b,
                                                  (short)0, acc, false, false);
  }
  const bf16_t* arow2 = A2 + (size_t)rowA * lda2;
  for (int kt = 0; kt < k2t; ++kt) {
    v16bf a = load_a_frag(arow2, kt * 32 + koff);
    v16bf b = *(const v16bf*)(bptr);
    bptr += bstep;
    acc = __builtin_amdgcn_wmma_f32_16x16x32_bf16(false, a, false, b,
                                                  (short)0, acc, false, false);
  }

  const float bv = bias[nt * 16 + laneM];
#pragma unroll
  for (int r = 0; r < 8; ++r) {               // C layout: VGPR r, lanes0-15 M=r,
    int row = mt * 16 + r + hi * 8;           // lanes16-31 M=r+8, N = lane&15
    if (row < M) C[(size_t)row * ldc + nt * 16 + laneM] = acc[r] + bv;
  }
}

// ---------------------------------------------------------------------------
// Joint kernel: out[row, :] = tanh(enc[b,t,:] + dec[b,u,:]) @ W_out^T + b_out
// row = ((b*T)+t)*U + u, 16 rows per block.  16 waves/block, 4 N-tiles/wave
// => full ODIM=1024.  The tanh'd A tile is built once in LDS in fragment
// layout (reused by all 64 N-tiles), B fragments stream from L2 w/ prefetch.
// Fill phase: with 512 threads and stride-512, every per-thread index except
// kt is loop-invariant -> hoisted.
// ---------------------------------------------------------------------------
__global__ __launch_bounds__(512) void joint_wmma_kernel(
    const float* __restrict__ encp, const float* __restrict__ decp,
    const bf16_t* __restrict__ Bout, const float* __restrict__ bout,
    float* __restrict__ out) {
  __shared__ bf16_t Alds[JKT * 512];
  const int tid = threadIdx.x;
  const int m   = blockIdx.x;

  {
    const int ln    = tid >> 4;        // 0..31 (fragment lane)
    const int j     = tid & 15;        // element within lane
    const int laneM = ln & 15;
    const int rowg  = m * 16 + laneM;
    const int b     = rowg / (TT * UU);
    const int rem   = rowg - b * (TT * UU);
    const int t     = rem / UU;
    const int u     = rem - t * UU;
    const int kofs  = (ln < 16) ? j : 16 + j;
    const float* __restrict__ erow = encp + (size_t)(b * TT + t) * JOINTD;
    const float* __restrict__ drow = decp + (size_t)(b * UU + u) * JOINTD;
#pragma unroll 4
    for (int kt = 0; kt < JKT; ++kt) {
      int k = kt * 32 + kofs;
      Alds[kt * 512 + tid] = (bf16_t)fast_tanh(erow[k] + drow[k]);
    }
  }
  __syncthreads();

  const int lane = tid & 31;
  const int wid  = tid >> 5;
  v8f z = {0.f, 0.f, 0.f, 0.f, 0.f, 0.f, 0.f, 0.f};
  v8f acc0 = z, acc1 = z, acc2 = z, acc3 = z;

  const bf16_t* bp = Bout + (((size_t)wid * 4) * 32 + lane) * 16;
  const size_t  bstep = (size_t)64 * 512;       // next kt (64 N-tiles)
  for (int kt = 0; kt < JKT; ++kt) {
    v16bf a = *(const v16bf*)(&Alds[(kt * 32 + lane) * 16]);
    if (kt + 1 < JKT) __builtin_prefetch(bp + bstep, 0, 0);
    v16bf b0 = *(const v16bf*)(bp);
    v16bf b1 = *(const v16bf*)(bp + 1 * 32 * 16);
    v16bf b2 = *(const v16bf*)(bp + 2 * 32 * 16);
    v16bf b3 = *(const v16bf*)(bp + 3 * 32 * 16);
    bp += bstep;
    acc0 = __builtin_amdgcn_wmma_f32_16x16x32_bf16(false, a, false, b0, (short)0, acc0, false, false);
    acc1 = __builtin_amdgcn_wmma_f32_16x16x32_bf16(false, a, false, b1, (short)0, acc1, false, false);
    acc2 = __builtin_amdgcn_wmma_f32_16x16x32_bf16(false, a, false, b2, (short)0, acc2, false, false);
    acc3 = __builtin_amdgcn_wmma_f32_16x16x32_bf16(false, a, false, b3, (short)0, acc3, false, false);
  }

  const int laneM = lane & 15, hi = lane >> 4;
  v8f accs[4] = {acc0, acc1, acc2, acc3};
#pragma unroll
  for (int i = 0; i < 4; ++i) {
    int n0 = (wid * 4 + i) * 16;
    float bv = bout[n0 + laneM];
#pragma unroll
    for (int r = 0; r < 8; ++r) {
      int rowg = m * 16 + r + hi * 8;
      out[(size_t)rowg * ODIM + n0 + laneM] = accs[i][r] + bv;
    }
  }
}

// ---------------------------------------------------------------------------
// LSTM gate nonlinearity (torch order i,f,g,o). Writes bf16 h into the
// recurrent concat buffer, and optionally into h_dec at row (b*U + u).
// ---------------------------------------------------------------------------
__global__ void lstm_gate_kernel(const float* __restrict__ gates,
                                 float* __restrict__ c,
                                 bf16_t* __restrict__ h_bf, int hStride,
                                 bf16_t* __restrict__ hdec, int u) {
  int idx = blockIdx.x * blockDim.x + threadIdx.x;
  if (idx >= BB * DUNITS) return;
  int r  = idx >> 10;
  int hh = idx & (DUNITS - 1);
  const float* g = gates + (size_t)r * (4 * DUNITS);
  float ig = fast_sigmoid(g[hh]);
  float fg = fast_sigmoid(g[DUNITS + hh]);
  float gg = fast_tanh  (g[2 * DUNITS + hh]);
  float og = fast_sigmoid(g[3 * DUNITS + hh]);
  float cn = fg * c[idx] + ig * gg;
  c[idx] = cn;
  float hn = og * fast_tanh(cn);
  h_bf[(size_t)r * hStride + hh] = (bf16_t)hn;
  if (hdec) hdec[((size_t)(r * UU + u)) * DUNITS + hh] = (bf16_t)hn;
}

// Embedding gather + cast, layout (U, B, EMBED)
__global__ void embed_kernel(const float* __restrict__ embW,
                             const int* __restrict__ ys,
                             bf16_t* __restrict__ eys) {
  int idx = blockIdx.x * blockDim.x + threadIdx.x;
  if (idx >= UU * BB * EMBED) return;
  int e  = idx & (EMBED - 1);
  int rb = idx >> 9;
  int u  = rb >> 3;
  int b  = rb & 7;
  int tok = ys[b * UU + u];
  eys[idx] = (bf16_t)embW[(size_t)tok * EMBED + e];
}

__global__ void cast_f32_bf16_kernel(const float* __restrict__ src,
                                     bf16_t* __restrict__ dst, int n) {
  int i = blockIdx.x * blockDim.x + threadIdx.x;
  if (i < n) dst[i] = (bf16_t)src[i];
}

__global__ void init_state_kernel(bf16_t* __restrict__ hcat,
                                  float* __restrict__ c0,
                                  float* __restrict__ c1) {
  int i = blockIdx.x * blockDim.x + threadIdx.x;
  if (i < BB * 2 * DUNITS) hcat[i] = (bf16_t)0.0f;
  if (i < BB * DUNITS) { c0[i] = 0.0f; c1[i] = 0.0f; }
}

__global__ void bias_sum_kernel(const float* __restrict__ a, const float* __restrict__ b,
                                const float* __restrict__ cc, const float* __restrict__ d,
                                float* __restrict__ s0, float* __restrict__ s1) {
  int i = blockIdx.x * blockDim.x + threadIdx.x;
  if (i < 4 * DUNITS) s0[i] = a[i] + b[i];
  else if (i < 8 * DUNITS) { int j = i - 4 * DUNITS; s1[j] = cc[j] + d[j]; }
}

// ---------------------------------------------------------------------------
extern "C" void kernel_launch(void* const* d_in, const int* in_sizes, int n_in,
                              void* d_out, int out_size, void* d_ws, size_t ws_size,
                              hipStream_t stream) {
  (void)in_sizes; (void)n_in; (void)out_size; (void)ws_size;
  const float* hs_pad = (const float*)d_in[0];
  const int*   ys     = (const int*)d_in[1];
  const float* embW   = (const float*)d_in[2];
  const float* W_ih0  = (const float*)d_in[3];
  const float* W_hh0  = (const float*)d_in[4];
  const float* b_ih0  = (const float*)d_in[5];
  const float* b_hh0  = (const float*)d_in[6];
  const float* W_ih1  = (const float*)d_in[7];
  const float* W_hh1  = (const float*)d_in[8];
  const float* b_ih1  = (const float*)d_in[9];
  const float* b_hh1  = (const float*)d_in[10];
  const float* W_enc  = (const float*)d_in[11];
  const float* b_enc  = (const float*)d_in[12];
  const float* W_dec  = (const float*)d_in[13];
  const float* b_dec  = (const float*)d_in[14];
  const float* W_out  = (const float*)d_in[15];
  const float* b_out  = (const float*)d_in[16];
  float* out = (float*)d_out;

  char* p = (char*)d_ws;
  auto alloc = [&](size_t bytes) -> char* {
    char* r = p; p += (bytes + 255) & ~(size_t)255; return r;
  };
  bf16_t* hs_bf   = (bf16_t*)alloc((size_t)BB * TT * EPROJS * 2);   // (B*T, 512)
  bf16_t* eys_bf  = (bf16_t*)alloc((size_t)UU * BB * EMBED * 2);    // (U*B, 512)
  bf16_t* hdec_bf = (bf16_t*)alloc((size_t)BB * UU * DUNITS * 2);   // (B*U, 1024)
  bf16_t* hcat    = (bf16_t*)alloc((size_t)BB * 2 * DUNITS * 2);    // [h0|h1] (8,2048)
  float*  c0      = (float*)alloc((size_t)BB * DUNITS * 4);
  float*  c1      = (float*)alloc((size_t)BB * DUNITS * 4);
  float*  bsum0   = (float*)alloc((size_t)4 * DUNITS * 4);
  float*  bsum1   = (float*)alloc((size_t)4 * DUNITS * 4);
  float*  gates0  = (float*)alloc((size_t)BB * 4 * DUNITS * 4);
  float*  gates1  = (float*)alloc((size_t)BB * 4 * DUNITS * 4);
  float*  encp    = (float*)alloc((size_t)BB * TT * JOINTD * 4);
  float*  decp    = (float*)alloc((size_t)BB * UU * JOINTD * 4);
  bf16_t* Bp_enc  = (bf16_t*)alloc((size_t)EPROJS * JOINTD * 2);    // K=512,  N=640
  bf16_t* Bp_dec  = (bf16_t*)alloc((size_t)DUNITS * JOINTD * 2);    // K=1024, N=640
  bf16_t* Bp_out  = (bf16_t*)alloc((size_t)JOINTD * ODIM * 2);      // K=640,  N=1024
  bf16_t* Bp_l0   = (bf16_t*)alloc((size_t)(EMBED + DUNITS) * 4 * DUNITS * 2); // K=1536,N=4096
  bf16_t* Bp_l1   = (bf16_t*)alloc((size_t)(2 * DUNITS) * 4 * DUNITS * 2);     // K=2048,N=4096

  // ---- setup / cast / pack ------------------------------------------------
  {
    int n = BB * TT * EPROJS;
    cast_f32_bf16_kernel<<<(n + 255) / 256, 256, 0, stream>>>(hs_pad, hs_bf, n);
  }
  embed_kernel<<<(UU * BB * EMBED + 255) / 256, 256, 0, stream>>>(embW, ys, eys_bf);
  init_state_kernel<<<(BB * 2 * DUNITS + 255) / 256, 256, 0, stream>>>(hcat, c0, c1);
  bias_sum_kernel<<<(8 * DUNITS + 255) / 256, 256, 0, stream>>>(
      b_ih0, b_hh0, b_ih1, b_hh1, bsum0, bsum1);

  auto pack = [&](const float* W, bf16_t* dst, int K, int NT, int ktOff) {
    int ktLocal = K / 32;
    int total = ktLocal * NT * 512;
    pack_b_kernel<<<(total + 255) / 256, 256, 0, stream>>>(W, dst, K, ktLocal, NT, ktOff);
  };
  pack(W_enc, Bp_enc, EPROJS, JOINTD / 16, 0);    // 640/16 = 40 N-tiles
  pack(W_dec, Bp_dec, DUNITS, JOINTD / 16, 0);
  pack(W_out, Bp_out, JOINTD, ODIM / 16, 0);      // 64 N-tiles
  pack(W_ih0, Bp_l0, EMBED,  256, 0);             // K-tiles 0..15   (embed part)
  pack(W_hh0, Bp_l0, DUNITS, 256, 16);            // K-tiles 16..47  (h0 part)
  pack(W_ih1, Bp_l1, DUNITS, 256, 0);             // K-tiles 0..31   (h0 part)
  pack(W_hh1, Bp_l1, DUNITS, 256, 32);            // K-tiles 32..63  (h1 part)

  // ---- sequential 2-layer LSTM over U steps -------------------------------
  for (int u = 0; u < UU; ++u) {
    // layer0: gates0 = [ey_u | h0] @ [W_ih0|W_hh0]^T + (b_ih0+b_hh0)
    gemm_wmma_bf16<<<dim3(1, 64), 128, 0, stream>>>(
        eys_bf + (size_t)u * BB * EMBED, EMBED, EMBED / 32,
        hcat, 2 * DUNITS, DUNITS / 32,
        Bp_l0, 256, bsum0, gates0, 4 * DUNITS, BB);
    lstm_gate_kernel<<<(BB * DUNITS + 255) / 256, 256, 0, stream>>>(
        gates0, c0, hcat /* h0 slot */, 2 * DUNITS, nullptr, u);
    // layer1: gates1 = [h0 | h1] @ [W_ih1|W_hh1]^T + (b_ih1+b_hh1)
    gemm_wmma_bf16<<<dim3(1, 64), 128, 0, stream>>>(
        hcat, 2 * DUNITS, (2 * DUNITS) / 32,
        hcat, 2 * DUNITS, 0,
        Bp_l1, 256, bsum1, gates1, 4 * DUNITS, BB);
    lstm_gate_kernel<<<(BB * DUNITS + 255) / 256, 256, 0, stream>>>(
        gates1, c1, hcat + DUNITS /* h1 slot */, 2 * DUNITS, hdec_bf, u);
  }

  // ---- projections --------------------------------------------------------
  gemm_wmma_bf16<<<dim3((BB * TT) / 16, 10), 128, 0, stream>>>(
      hs_bf, EPROJS, EPROJS / 32, hs_bf, EPROJS, 0,
      Bp_enc, JOINTD / 16, b_enc, encp, JOINTD, BB * TT);
  gemm_wmma_bf16<<<dim3((BB * UU) / 16, 10), 128, 0, stream>>>(
      hdec_bf, DUNITS, DUNITS / 32, hdec_bf, DUNITS, 0,
      Bp_dec, JOINTD / 16, b_dec, decp, JOINTD, BB * UU);

  // ---- joint: tanh(enc+dec) @ W_out^T + b_out  (the 63 GFLOP whale) -------
  joint_wmma_kernel<<<dim3((BB * TT * UU) / 16), 512, 0, stream>>>(
      encp, decp, Bp_out, b_out, out);
}
